// VGG16_TTConv_41137196761157
// MI455X (gfx1250) — matrix operand order repaired and checked by reading, not compiled, not run on hardware
//
#include <hip/hip_runtime.h>

// ---------------------------------------------------------------------------
// Types for CDNA5 WMMA (wave32): A/B are 16 bf16 per lane, C/D are 8 f32/lane.
// ---------------------------------------------------------------------------
typedef __attribute__((ext_vector_type(16))) __bf16          v16bf;
typedef __attribute__((ext_vector_type(16))) unsigned short  us16;
typedef __attribute__((ext_vector_type(8)))  float           v8f;

struct __align__(16) U4 { unsigned x, y, z, w; };
struct __align__(16) Q2 { U4 a, b; };   // 32 bytes == one 16-element bf16 frag

__device__ __forceinline__ v16bf mkfrag(U4 a, U4 b) {
    Q2 q; q.a = a; q.b = b;
    return __builtin_bit_cast(v16bf, q);
}
__device__ __forceinline__ float bf2f(unsigned short u) {
    return __uint_as_float(((unsigned int)u) << 16);
}
__device__ __forceinline__ unsigned short f2bf(float f) {
    unsigned int u = __float_as_uint(f);
    u += 0x7fffu + ((u >> 16) & 1u);   // round-to-nearest-even
    return (unsigned short)(u >> 16);
}

// ---------------------------------------------------------------------------
// NCHW fp32 -> NHWC bf16
// ---------------------------------------------------------------------------
__global__ void k_nchw_to_nhwc_bf16(const float* __restrict__ in,
                                    unsigned short* __restrict__ out,
                                    int NB, int C, int H, int W) {
    long long total = (long long)NB * C * H * W;
    long long i = (long long)blockIdx.x * blockDim.x + threadIdx.x;
    if (i >= total) return;
    long long t = i;
    int c = (int)(t % C); t /= C;
    int x = (int)(t % W); t /= W;
    int y = (int)(t % H); t /= H;
    int n = (int)t;
    float v = in[(((long long)n * C + c) * H + y) * W + x];
    out[i] = f2bf(v);
}

// ---------------------------------------------------------------------------
// One TT contraction step:  out[N*n, M*m, B] = sum_a Wp[N,M,a] * G[a,n,m,B]
// (einsum 'NMa,anmb->NnMmb' of the reference, flattened)
// ---------------------------------------------------------------------------
__global__ void k_tt_contract(const float* __restrict__ Wp,
                              const float* __restrict__ G,
                              float* __restrict__ out,
                              int N, int M, int A, int n, int m, int B) {
    long long total = (long long)N * n * M * m * B;
    long long i = (long long)blockIdx.x * blockDim.x + threadIdx.x;
    if (i >= total) return;
    long long t = i;
    int bi  = (int)(t % B); t /= B;
    int col = (int)(t % ((long long)M * m)); t /= ((long long)M * m);
    int row = (int)t;
    int Ni = row / n, ni = row - Ni * n;
    int Mi = col / m, mi = col - Mi * m;
    const float* wp = Wp + ((long long)Ni * M + Mi) * A;
    const float* g  = G + (((long long)ni * m + mi) * B + bi);
    long long gs = (long long)n * m * B;
    float s = 0.f;
    for (int a = 0; a < A; ++a) s += wp[a] * g[(long long)a * gs];
    out[i] = s;
}

// fp32 [K, C] -> bf16 transposed [C, K]  (weight layout for direct B frags)
__global__ void k_f32_to_bf16_T(const float* __restrict__ in,
                                unsigned short* __restrict__ out,
                                int K, int C) {
    long long total = (long long)K * C;
    long long i = (long long)blockIdx.x * blockDim.x + threadIdx.x;
    if (i >= total) return;
    int k  = (int)(i % K);
    int co = (int)(i / K);
    out[i] = f2bf(in[(long long)k * C + co]);
}

__global__ void k_f32_to_bf16(const float* __restrict__ in,
                              unsigned short* __restrict__ out, long long total) {
    long long i = (long long)blockIdx.x * blockDim.x + threadIdx.x;
    if (i >= total) return;
    out[i] = f2bf(in[i]);
}

// ---------------------------------------------------------------------------
// Conv3x3 SAME, implicit-GEMM with V_WMMA_F32_16X16X32_BF16.
// Block = 128 threads (4 waves); tile = 64 pixels x 64 cout.
// Activations NHWC bf16; weights bf16 TRANSPOSED [Cout, 9*Cin].
// A staged into LDS with GLOBAL_LOAD_ASYNC_TO_LDS_B128 (ASYNCcnt path,
// bypasses VGPRs); B read directly from global (L2-resident weights).
// Fast path (Cin = pow2 >= 32): all-vector b128 traffic, no K guards.
// ---------------------------------------------------------------------------
__global__ void __launch_bounds__(128)
k_conv3x3_wmma(const unsigned short* __restrict__ in,
               const unsigned short* __restrict__ wT,
               const float* __restrict__ bias,
               unsigned short* __restrict__ out,
               int NB, int H, int W, int Cin, int Cout) {
    __shared__ unsigned short lA[64][40];   // pitch 80B (16B multiple)

    const int P    = NB * H * W;
    const int Kdim = 9 * Cin;
    const int p0   = blockIdx.x * 64;
    const int coT  = blockIdx.y * 64;
    const int t    = threadIdx.x;
    const int lane = t & 31;
    const int wv   = t >> 5;
    const int rlo  = lane & 15;
    const int hi   = (lane >> 4) & 1;

    // This thread's half-row of the A tile
    const int pl  = t >> 1;           // pixel row 0..63
    const int kk0 = (t & 1) * 16;     // k offset 0 or 16
    int pix = p0 + pl;
    bool pvalid = pix < P;
    int pp  = pvalid ? pix : 0;
    int n   = pp / (H * W);
    int rem = pp - n * (H * W);
    int y   = rem / W;
    int x   = rem - y * W;
    const int pixAddr = ((n * H + y) * W + x) * Cin;  // fits int (< 2^29)

    // LDS byte offset of this thread's staging slot (low 32 bits of the
    // generic shared address are the LDS offset on AMDGPU).
    const unsigned ldsOff = (unsigned)(size_t)(void*)&lA[pl][kk0];

    const int co = coT + wv * 16 + rlo;     // Cout multiple of 64 -> valid
    const unsigned short* wrow = wT + (long long)co * Kdim;

    const bool fast  = (Cin >= 32) && ((Cin & (Cin - 1)) == 0);
    const int cinLog = 31 - __clz(Cin);

    v8f zero = {0.f, 0.f, 0.f, 0.f, 0.f, 0.f, 0.f, 0.f};
    v8f acc[4];
    acc[0] = zero; acc[1] = zero; acc[2] = zero; acc[3] = zero;
    U4 z4; z4.x = 0; z4.y = 0; z4.z = 0; z4.w = 0;

    for (int kb = 0; kb < Kdim; kb += 32) {
        // ---------------- stage A tile into LDS ----------------
        if (fast) {
            // whole 32-k block lies inside one (kh,kw) tap
            int s   = kb >> cinLog;
            int ci0 = (kb & (Cin - 1)) + kk0;
            int kh  = s / 3, kw = s - 3 * kh;
            int yy  = y + kh - 1, xx = x + kw - 1;
            bool inb = pvalid && ((unsigned)yy < (unsigned)H) &&
                                 ((unsigned)xx < (unsigned)W);
            int soff = ((kh - 1) * W + (kw - 1)) * Cin;  // uniform
            if (inb) {
                unsigned long long ga =
                    (unsigned long long)(size_t)(in + pixAddr + soff + ci0);
                // 32B strip: global -> LDS without touching VGPRs (ASYNCcnt)
                asm volatile("global_load_async_to_lds_b128 %0, %1, off"
                             :: "v"(ldsOff), "v"(ga) : "memory");
                asm volatile("global_load_async_to_lds_b128 %0, %1, off offset:16"
                             :: "v"(ldsOff), "v"(ga) : "memory");
            } else {
                *(U4*)&lA[pl][kk0]     = z4;   // zero padding / tail pixels
                *(U4*)&lA[pl][kk0 + 8] = z4;
            }
            // drain this wave's async copies before the barrier
            asm volatile("s_wait_asynccnt 0x0" ::: "memory");
        } else {
            // generic path (conv0: Cin=3, Kdim=27)
            int k  = kb + kk0;
            int s  = k / Cin;
            int ci = k - s * Cin;
            for (int i = 0; i < 16; ++i) {
                unsigned short v = 0;
                if (pvalid && (k + i) < Kdim) {
                    int kh = s / 3, kw = s - kh * 3;
                    int yy = y + kh - 1, xx = x + kw - 1;
                    if (yy >= 0 && yy < H && xx >= 0 && xx < W)
                        v = in[(((long long)n * H + yy) * W + xx) * Cin + ci];
                }
                lA[pl][kk0 + i] = v;
                ++ci; if (ci == Cin) { ci = 0; ++s; }
            }
        }
        __syncthreads();

        // ---------------- B fragment: direct from global (L2) ----------------
        v16bf bfrag;
        if (fast) {
            const U4* wp = (const U4*)(wrow + kb + hi * 16);
            bfrag = mkfrag(wp[0], wp[1]);
            if (kb + 32 < Kdim)
                __builtin_prefetch(wrow + kb + 32, 0, 1);
        } else {
            us16 bvu;
            for (int j = 0; j < 16; ++j) {
                int k = kb + hi * 16 + j;
                bvu[j] = (k < Kdim) ? wrow[k] : (unsigned short)0;
            }
            bfrag = __builtin_bit_cast(v16bf, bvu);
        }

        // ---------------- A fragments (2x ds_load_b128 each) + WMMA ----------
#pragma unroll
        for (int mt = 0; mt < 4; ++mt) {
            const U4* rowp = (const U4*)&lA[mt * 16 + rlo][0];
            v16bf afrag = mkfrag(rowp[2 * hi], rowp[2 * hi + 1]);
            acc[mt] = __builtin_amdgcn_wmma_f32_16x16x32_bf16(
                false, afrag, false, bfrag, (short)0, acc[mt], false, false);
        }
        __syncthreads();
    }

    // ---- epilogue: bias + ReLU, store bf16 NHWC ----
    float bco = bias[co];
#pragma unroll
    for (int mt = 0; mt < 4; ++mt) {
#pragma unroll
        for (int v = 0; v < 8; ++v) {
            int pixo = p0 + mt * 16 + hi * 8 + v;
            if (pixo < P) {
                float val = acc[mt][v] + bco;
                val = val > 0.f ? val : 0.f;
                out[(long long)pixo * Cout + co] = f2bf(val);
            }
        }
    }
}

// ---------------------------------------------------------------------------
// MaxPool 2x2 stride 2, NHWC bf16
// ---------------------------------------------------------------------------
__global__ void k_maxpool2(const unsigned short* __restrict__ in,
                           unsigned short* __restrict__ out,
                           int NB, int H, int W, int C) {
    int Ho = H >> 1, Wo = W >> 1;
    long long total = (long long)NB * Ho * Wo * C;
    long long i = (long long)blockIdx.x * blockDim.x + threadIdx.x;
    if (i >= total) return;
    long long t = i;
    int c = (int)(t % C); t /= C;
    int x = (int)(t % Wo); t /= Wo;
    int y = (int)(t % Ho); t /= Ho;
    int n = (int)t;
    const unsigned short* p = in + (((long long)n * H + 2 * y) * W + 2 * x) * C + c;
    long long rs = (long long)W * C;
    float a = bf2f(p[0]), b = bf2f(p[C]);
    float d0 = bf2f(p[rs]), d1 = bf2f(p[rs + C]);
    out[i] = f2bf(fmaxf(fmaxf(a, b), fmaxf(d0, d1)));
}

// ---------------------------------------------------------------------------
// Global average pool: NHWC bf16 [NB,HW,C] -> fp32 [NB,C]
// ---------------------------------------------------------------------------
__global__ void k_avgpool(const unsigned short* __restrict__ in,
                          float* __restrict__ out, int NB, int HW, int C) {
    int i = blockIdx.x * blockDim.x + threadIdx.x;
    if (i >= NB * C) return;
    int c = i % C, n = i / C;
    float s = 0.f;
    for (int p = 0; p < HW; ++p)
        s += bf2f(in[((long long)n * HW + p) * C + c]);
    out[i] = s / (float)HW;
}

// ---------------------------------------------------------------------------
// FC layer with WMMA: x[16,Fin] bf16 @ wT[Fout,Fin] bf16 (transposed) + bias.
// One wave per 16 output features; all-vector b128 global loads.
// relu_bf16=1 -> ReLU + bf16 out, 0 -> fp32 out (final layer).
// ---------------------------------------------------------------------------
__global__ void __launch_bounds__(32)
k_linear_wmma(const unsigned short* __restrict__ xin,
              const unsigned short* __restrict__ wT,
              const float* __restrict__ bias,
              void* __restrict__ outp,
              int Fin, int Fout, int relu_bf16) {
    int lane = threadIdx.x & 31;
    int rlo = lane & 15;
    int hi = (lane >> 4) & 1;
    int co = blockIdx.x * 16 + rlo;
    bool cvalid = co < Fout;
    const unsigned short* wrow = wT + (long long)(cvalid ? co : 0) * Fin;
    const unsigned short* xrow = xin + (long long)rlo * Fin;

    U4 z4; z4.x = 0; z4.y = 0; z4.z = 0; z4.w = 0;
    v8f acc = {0.f, 0.f, 0.f, 0.f, 0.f, 0.f, 0.f, 0.f};
    for (int kb = 0; kb < Fin; kb += 32) {
        const U4* xp = (const U4*)(xrow + kb);          // Fin % 32 == 0
        v16bf af = mkfrag(xp[2 * hi], xp[2 * hi + 1]);
        U4 b0 = z4, b1 = z4;
        if (cvalid) {
            const U4* wp = (const U4*)(wrow + kb + hi * 16);
            b0 = wp[0]; b1 = wp[1];
        }
        v16bf bf = mkfrag(b0, b1);
        acc = __builtin_amdgcn_wmma_f32_16x16x32_bf16(
            false, af, false, bf, (short)0, acc, false, false);
    }
    if (cvalid) {
        float bc = bias[co];
#pragma unroll
        for (int v = 0; v < 8; ++v) {
            int b = hi * 8 + v;           // batch row
            float val = acc[v] + bc;
            if (relu_bf16) {
                val = fmaxf(val, 0.f);
                ((unsigned short*)outp)[(long long)b * Fout + co] = f2bf(val);
            } else {
                ((float*)outp)[(long long)b * Fout + co] = val;
            }
        }
    }
}

// ---------------------------------------------------------------------------
// Host orchestration
// ---------------------------------------------------------------------------
extern "C" void kernel_launch(void* const* d_in, const int* in_sizes, int n_in,
                              void* d_out, int out_size, void* d_ws, size_t ws_size,
                              hipStream_t stream) {
    (void)in_sizes; (void)n_in; (void)out_size; (void)ws_size;

    static const int NCONV = 13, NLIN = 3;
    static const int convD[NCONV] = {2,4,4,4,4,4,4,4,4,4,4,4,4};
    // n[0] already folds K*K=9
    static const int convN[NCONV][4] = {
        {9,3,1,1},{9,4,4,4},{9,4,4,4},{18,4,4,4},{18,4,4,4},{36,4,4,4},{36,4,4,4},
        {36,4,4,4},{72,4,4,4},{72,4,4,4},{72,4,4,4},{72,4,4,4},{72,4,4,4}};
    static const int convM[NCONV][4] = {
        {8,8,1,1},{1,4,4,4},{2,4,4,4},{2,4,4,4},{4,4,4,4},{4,4,4,4},{4,4,4,4},
        {8,4,4,4},{8,4,4,4},{8,4,4,4},{8,4,4,4},{8,4,4,4},{8,4,4,4}};
    static const int convR[NCONV][5] = {
        {1,9,1,1,1},{1,9,9,9,1},{1,9,9,9,1},{1,9,9,9,1},{1,9,9,9,1},{1,9,9,9,1},
        {1,9,9,9,1},{1,9,9,9,1},{1,9,9,9,1},{1,9,9,9,1},{1,9,9,9,1},{1,9,9,9,1},
        {1,9,9,9,1}};
    static const int convCin [NCONV] = {3,64,64,128,128,256,256,256,512,512,512,512,512};
    static const int convCout[NCONV] = {64,64,128,128,256,256,256,512,512,512,512,512,512};
    static const bool poolAfter[NCONV] =
        {false,true,false,true,false,false,true,false,false,true,false,false,true};

    static const int linN[NLIN][4] = {{4,4,4,8},{4,4,4,64},{4,4,4,64}};
    static const int linM[NLIN][4] = {{4,4,4,64},{4,4,4,64},{2,2,5,5}};
    static const int linR[NLIN][5] = {{1,9,9,9,1},{1,9,9,9,1},{1,11,11,11,1}};
    static const int linFin [NLIN] = {512,4096,4096};
    static const int linFout[NLIN] = {4096,4096,100};

    // ---- input pointer walk (setup_inputs dict insertion order) ----
    const float* coreP[16][4];
    const float* biasP[16];
    int ip = 1;  // d_in[0] = x
    for (int L = 0; L < NCONV; ++L) {
        int d = convD[L];
        for (int i = 0; i < d; ++i) coreP[L][i] = (const float*)d_in[ip + i];
        biasP[L] = (const float*)d_in[ip + d];
        ip += d + 1;
    }
    for (int L = 0; L < NLIN; ++L) {
        for (int i = 0; i < 4; ++i) coreP[NCONV + L][i] = (const float*)d_in[ip + i];
        biasP[NCONV + L] = (const float*)d_in[ip + 4];
        ip += 5;
    }

    // ---- workspace carve ----
    char* ws = (char*)d_ws;
    size_t off = 0;
    auto carve = [&](size_t bytes) -> void* {
        void* p = ws + off;
        off += (bytes + 255) & ~(size_t)255;
        return p;
    };
    unsigned short* actA = (unsigned short*)carve((size_t)16 * 224 * 224 * 64 * 2);
    unsigned short* actB = (unsigned short*)carve((size_t)16 * 224 * 224 * 64 * 2);
    unsigned short* wBuf = (unsigned short*)carve((size_t)4096 * 4096 * 2);
    float* s0            = (float*)carve((size_t)4096 * 4096 * 4);     // 67 MB
    float* s1            = (float*)carve((size_t)2 * 1024 * 1024 * 4); // 8 MB
    float* pooled        = (float*)carve((size_t)16 * 512 * 4);
    unsigned short* linIn = (unsigned short*)carve((size_t)16 * 512 * 2);
    unsigned short* linA  = (unsigned short*)carve((size_t)16 * 4096 * 2);
    unsigned short* linB  = (unsigned short*)carve((size_t)16 * 4096 * 2);

    // ---- build full weight matrix from TT cores -> wBuf (bf16, transposed) --
    auto buildW = [&](int li, const int* nA, const int* mA, const int* rA, int d) {
        const float* cur = coreP[li][0];   // [n0, m0, r1] (r0 == 1)
        int N = nA[0], M = mA[0];
        for (int i = 1; i < d; ++i) {
            int A = rA[i], n = nA[i], m = mA[i], B = rA[i + 1];
            float* op = (i & 1) ? s0 : s1;
            long long total = (long long)N * n * M * m * B;
            int blocks = (int)((total + 255) / 256);
            k_tt_contract<<<blocks, 256, 0, stream>>>(cur, coreP[li][i], op,
                                                      N, M, A, n, m, B);
            cur = op; N *= n; M *= m;
        }
        long long total = (long long)N * M;
        k_f32_to_bf16_T<<<(unsigned)((total + 255) / 256), 256, 0, stream>>>(
            cur, wBuf, N, M);   // [N=Kdim, M=Cout] -> bf16 [Cout, Kdim]
    };

    // ---- stage input NCHW fp32 -> NHWC bf16 ----
    {
        long long total = (long long)16 * 3 * 224 * 224;
        k_nchw_to_nhwc_bf16<<<(unsigned)((total + 255) / 256), 256, 0, stream>>>(
            (const float*)d_in[0], actA, 16, 3, 224, 224);
    }

    // ---- conv stack ----
    unsigned short* cur = actA;
    unsigned short* oth = actB;
    int H = 224, W = 224;
    for (int L = 0; L < NCONV; ++L) {
        buildW(L, convN[L], convM[L], convR[L], convD[L]);
        int Cin = convCin[L], Cout = convCout[L];
        int P = 16 * H * W;
        dim3 grid((P + 63) / 64, Cout / 64);
        k_conv3x3_wmma<<<grid, 128, 0, stream>>>(cur, wBuf, biasP[L], oth,
                                                 16, H, W, Cin, Cout);
        if (poolAfter[L]) {
            int Ho = H / 2, Wo = W / 2;
            long long total = (long long)16 * Ho * Wo * Cout;
            k_maxpool2<<<(unsigned)((total + 255) / 256), 256, 0, stream>>>(
                oth, cur, 16, H, W, Cout);
            H = Ho; W = Wo;               // pooled result is in `cur`
        } else {
            unsigned short* tmp = cur; cur = oth; oth = tmp;
        }
    }

    // ---- global average pool + classifier ----
    k_avgpool<<<(16 * 512 + 255) / 256, 256, 0, stream>>>(cur, pooled, 16, 49, 512);
    k_f32_to_bf16<<<(16 * 512 + 255) / 256, 256, 0, stream>>>(pooled, linIn,
                                                              (long long)16 * 512);
    const unsigned short* lx = linIn;
    for (int L = 0; L < NLIN; ++L) {
        buildW(NCONV + L, linN[L], linM[L], linR[L], 4);
        int Fin = linFin[L], Fout = linFout[L];
        void* op; int relu;
        if (L == 0)      { op = linA;  relu = 1; }
        else if (L == 1) { op = linB;  relu = 1; }
        else             { op = d_out; relu = 0; }
        k_linear_wmma<<<(Fout + 15) / 16, 32, 0, stream>>>(lx, wBuf,
                                                           biasP[NCONV + L], op,
                                                           Fin, Fout, relu);
        lx = (L == 0) ? linA : linB;
    }
}
